// VectorQuantizer_15341623181536
// MI455X (gfx1250) — matrix operand order repaired and checked by reading, not compiled, not run on hardware
//
#include <hip/hip_runtime.h>
#include <hip/hip_bf16.h>

typedef __attribute__((ext_vector_type(16))) _Float16 v16h;
typedef __attribute__((ext_vector_type(8)))  float    v8f;
typedef int b128_t __attribute__((vector_size(16)));   // 4 x i32, matches builtin

#define NUM_CODES 4096
#define DIM       256
#define NROWS     65536            // 64*1024
#define TILE_CODES 16
#define KSTEPS    8                // 256 / 32
#define ROWS_PER_WAVE 16
#define ROWS_PER_BLOCK 128
#define Q_ELEMS   16777216         // 65536*256
#define GL_BLOCKS 16384            // Q_ELEMS/4/256

#define TILE_BYTES  (TILE_CODES * DIM * 2)   // 8192 B of f16 codes
#define TILE_STRIDE (TILE_BYTES + 64)        // + 64 B staged enorm tile

// ---- CDNA5 async global->LDS staging (ASYNCcnt path), with safe fallback ----
#if defined(__gfx1250__) && __has_builtin(__builtin_amdgcn_global_load_async_to_lds_b128)
#define USE_ASYNC_LDS 1
#define GASYNC_B128(gptr, lptr)                                                  \
    __builtin_amdgcn_global_load_async_to_lds_b128(                              \
        (__attribute__((address_space(1))) b128_t*)(gptr),                       \
        (__attribute__((address_space(3))) b128_t*)(lptr), 0, 0)
#else
#define USE_ASYNC_LDS 0
#endif

#if __has_builtin(__builtin_amdgcn_s_wait_asynccnt)
#define WAIT_ASYNC0() __builtin_amdgcn_s_wait_asynccnt(0)
#else
#define WAIT_ASYNC0() asm volatile("s_wait_asynccnt 0x0" ::: "memory")
#endif

// ---------------- Kernel 1: emb f32 -> f16, and ||e||^2 ----------------
__global__ __launch_bounds__(256) void vq_prep_kernel(
    const float* __restrict__ emb, _Float16* __restrict__ eh,
    float* __restrict__ enorm)
{
    const int wave = threadIdx.x >> 5;
    const int lane = threadIdx.x & 31;
    const int row  = blockIdx.x * 8 + wave;          // grid = 512 -> 4096 rows
    const float* e = emb + (size_t)row * DIM + lane * 8;

    union { _Float16 h[8]; uint4 u; } pk;
    float s = 0.0f;
#pragma unroll
    for (int i = 0; i < 8; ++i) {
        float v = e[i];
        s += v * v;
        pk.h[i] = (_Float16)v;
    }
    *(uint4*)(eh + (size_t)row * DIM + lane * 8) = pk.u;
#pragma unroll
    for (int m = 1; m < 32; m <<= 1) s += __shfl_xor(s, m, 32);
    if (lane == 0) enorm[row] = s;
}

// ---------------- Kernel 2: WMMA distance GEMM + argmin ----------------
__global__ __launch_bounds__(256) void vq_argmin_kernel(
    const float* __restrict__ x,       // [65536][256] f32
    const _Float16* __restrict__ eh,   // [4096][256] f16
    const float* __restrict__ enorm,   // [4096] f32
    int* __restrict__ idx_out,         // [65536] int  (ws)
    float* __restrict__ idxf_out)      // [65536] float (d_out tail)
{
    // 2 buffers x (8KB emb-f16 tile + 64B enorm tile)
    __shared__ __attribute__((aligned(16))) unsigned char smem[2][TILE_STRIDE];

    const int tid  = threadIdx.x;
    const int wave = tid >> 5;
    const int lane = tid & 31;
    const int n    = lane & 15;    // column / row-in-group
    const int hs   = lane >> 4;    // lane-half select

    const int row_base = blockIdx.x * ROWS_PER_BLOCK + wave * ROWS_PER_WAVE;

    // Load & convert A fragments once: lane holds row (row_base+n),
    // K chunks {k0+16*hs .. +8} and {k0+16*hs+8 .. +8} (contiguous, matches B).
    union AF { v16h h; } a[KSTEPS];
    const float* xrow = x + (size_t)(row_base + n) * DIM;
#pragma unroll
    for (int kk = 0; kk < KSTEPS; ++kk) {
        const int k0 = kk * 32;
        const float* c1 = xrow + k0 + 16 * hs;
        const float* c2 = xrow + k0 + 16 * hs + 8;
#pragma unroll
        for (int i = 0; i < 8; ++i) {
            a[kk].h[i]     = (_Float16)c1[i];
            a[kk].h[i + 8] = (_Float16)c2[i];
        }
    }

    float best[8];
    int   bidx[8];
#pragma unroll
    for (int r = 0; r < 8; ++r) { best[r] = 3.4e38f; bidx[r] = 0; }

    const int NTILES = NUM_CODES / TILE_CODES;    // 256

#if USE_ASYNC_LDS
    // ---- stage tile 0 into buf entirely via async global->LDS DMA ----
    {
        const unsigned char* g = (const unsigned char*)eh;       // tile 0
        unsigned char* b = smem[0];
        GASYNC_B128(g + tid * 16, b + tid * 16);
        GASYNC_B128(g + 4096 + tid * 16, b + 4096 + tid * 16);
        if (tid < 4)
            GASYNC_B128((const unsigned char*)enorm + tid * 16,
                        b + TILE_BYTES + tid * 16);
    }
    for (int t = 0; t < NTILES; ++t) {
        WAIT_ASYNC0();                 // my async writes to smem[t&1] landed
        __syncthreads();               // all waves' writes landed; prev reads done
        const int cur = t & 1;
        if (t + 1 < NTILES) {          // kick DMA for next tile into other buffer
            const unsigned char* g =
                (const unsigned char*)(eh + (size_t)(t + 1) * TILE_CODES * DIM);
            unsigned char* b = smem[cur ^ 1];
            GASYNC_B128(g + tid * 16, b + tid * 16);
            GASYNC_B128(g + 4096 + tid * 16, b + 4096 + tid * 16);
            if (tid < 4)
                GASYNC_B128((const unsigned char*)(enorm + (t + 1) * TILE_CODES)
                                + tid * 16,
                            b + TILE_BYTES + tid * 16);
        }
#else
    // ---- fallback: register double-buffer staging ----
    {
        const uint4* g = (const uint4*)eh;
        uint4* s = (uint4*)smem[0];
        s[tid]       = g[tid];
        s[tid + 256] = g[tid + 256];
        if (tid < 16) ((float*)(smem[0] + TILE_BYTES))[tid] = enorm[tid];
    }
    for (int t = 0; t < NTILES; ++t) {
        __syncthreads();
        const int cur = t & 1;
        uint4 p0, p1;
        float pe = 0.0f;
        const bool hasNext = (t + 1) < NTILES;
        if (hasNext) {
            const uint4* g = (const uint4*)(eh + (size_t)(t + 1) * TILE_CODES * DIM);
            p0 = g[tid];
            p1 = g[tid + 256];
            if (tid < 16) pe = enorm[(t + 1) * TILE_CODES + tid];
        }
#endif
        // ---- preload ALL B fragments, then run the full WMMA chain ----
        const _Float16* bt = (const _Float16*)smem[cur];
        union BF { v16h h; uint4 u[2]; } b[KSTEPS];
#pragma unroll
        for (int kk = 0; kk < KSTEPS; ++kk) {
            // B lane layout: lane holds 16 consecutive K of column n.
            const uint4* bp = (const uint4*)(bt + n * DIM + kk * 32 + 16 * hs);
            b[kk].u[0] = bp[0];
            b[kk].u[1] = bp[1];
        }
        v8f acc = {};
#pragma unroll
        for (int kk = 0; kk < KSTEPS; ++kk)
            acc = __builtin_amdgcn_wmma_f32_16x16x32_f16(
                false, a[kk].h, false, b[kk].h, (short)0, acc, false, false);

        const float en = ((const float*)(smem[cur] + TILE_BYTES))[n];  // from LDS
        const int code = t * TILE_CODES + n;
#pragma unroll
        for (int r = 0; r < 8; ++r) {
            // ||e||^2 - 2 x.e  (||x||^2 constant per row for argmin)
            float s = __builtin_fmaf(-2.0f, acc[r], en);
            if (s < best[r]) { best[r] = s; bidx[r] = code; }  // '<' keeps first hit
        }

#if !USE_ASYNC_LDS
        if (hasNext) {
            uint4* s = (uint4*)smem[cur ^ 1];
            s[tid]       = p0;
            s[tid + 256] = p1;
            if (tid < 16) ((float*)(smem[cur ^ 1] + TILE_BYTES))[tid] = pe;
        }
#endif
    }

    // Cross-lane argmin within each 16-lane group (rows r and r+8).
#pragma unroll
    for (int r = 0; r < 8; ++r) {
        float d = best[r];
        int   i = bidx[r];
#pragma unroll
        for (int m = 1; m < 16; m <<= 1) {
            float d2 = __shfl_xor(d, m, 32);
            int   i2 = __shfl_xor(i, m, 32);
            if (d2 < d || (d2 == d && i2 < i)) { d = d2; i = i2; }
        }
        if (n == 0) {
            const int row = row_base + r + 8 * hs;
            idx_out[row]  = i;
            idxf_out[row] = (float)i;
        }
    }
}

// ---------------- Kernel 3: gather quantized + per-block loss partials ----------------
__global__ __launch_bounds__(256) void vq_gather_loss_kernel(
    const float* __restrict__ x, const float* __restrict__ emb,
    const int* __restrict__ idx, float* __restrict__ qout,
    float* __restrict__ partials)
{
    const int gid  = blockIdx.x * 256 + threadIdx.x; // float4 index
    const int row  = gid >> 6;                       // 64 float4 per row
    const int d4   = gid & 63;
    const int code = idx[row];

    const float4 q  = ((const float4*)emb)[(size_t)code * 64 + d4];
    const float4 xv = ((const float4*)x)[gid];
    ((float4*)qout)[gid] = q;

    float dx = q.x - xv.x, dy = q.y - xv.y, dz = q.z - xv.z, dw = q.w - xv.w;
    float s = dx * dx + dy * dy + dz * dz + dw * dw;

#pragma unroll
    for (int m = 1; m < 32; m <<= 1) s += __shfl_xor(s, m, 32);

    __shared__ float red[8];
    const int wave = threadIdx.x >> 5;
    if ((threadIdx.x & 31) == 0) red[wave] = s;
    __syncthreads();
    if (threadIdx.x == 0) {
        float t = 0.0f;
        for (int i = 0; i < 8; ++i) t += red[i];   // fixed order -> deterministic
        partials[blockIdx.x] = t;
    }
}

// ---------------- Kernel 4: deterministic reduce -> loss ----------------
__global__ __launch_bounds__(256) void vq_loss_reduce_kernel(
    const float* __restrict__ partials, float* __restrict__ loss)
{
    float s = 0.0f;
    for (int i = 0; i < GL_BLOCKS / 256; ++i)      // fixed order
        s += partials[threadIdx.x + 256 * i];
    __shared__ float red[256];
    red[threadIdx.x] = s;
    __syncthreads();
    for (int off = 128; off > 0; off >>= 1) {
        if (threadIdx.x < off) red[threadIdx.x] += red[threadIdx.x + off];
        __syncthreads();
    }
    if (threadIdx.x == 0)
        *loss = red[0] * (1.25f / (float)Q_ELEMS); // q_loss + 0.25*e_loss
}

extern "C" void kernel_launch(void* const* d_in, const int* in_sizes, int n_in,
                              void* d_out, int out_size, void* d_ws, size_t ws_size,
                              hipStream_t stream) {
    (void)in_sizes; (void)n_in; (void)out_size; (void)ws_size;
    const float* x   = (const float*)d_in[0];   // [64,1024,256] f32
    const float* emb = (const float*)d_in[1];   // [4096,256]   f32

    float* out  = (float*)d_out;
    float* qout = out;                  // 16777216 quantized
    float* loss = out + Q_ELEMS;        // 1
    float* idxf = out + Q_ELEMS + 1;    // 65536 indices (as float)

    char* ws = (char*)d_ws;             // ~2.44 MB used
    _Float16* eh    = (_Float16*)ws;                               // 2,097,152 B
    float*    enorm = (float*)(ws + 2097152);                      //    16,384 B
    int*      idx   = (int*)(ws + 2097152 + 16384);                //   262,144 B
    float*    parts = (float*)(ws + 2097152 + 16384 + 262144);     //    65,536 B

    vq_prep_kernel<<<512, 256, 0, stream>>>(emb, eh, enorm);
    vq_argmin_kernel<<<NROWS / ROWS_PER_BLOCK, 256, 0, stream>>>(x, eh, enorm, idx, idxf);
    vq_gather_loss_kernel<<<GL_BLOCKS, 256, 0, stream>>>(x, emb, idx, qout, parts);
    vq_loss_reduce_kernel<<<1, 256, 0, stream>>>(parts, loss);
}
